// MambaLayer_cube_36326833389654
// MI455X (gfx1250) — matrix operand order repaired and checked by reading, not compiled, not run on hardware
//
#include <hip/hip_runtime.h>
#include <hip/hip_bf16.h>

typedef __attribute__((ext_vector_type(16))) _Float16 v16h;
typedef __attribute__((ext_vector_type(8)))  float    v8f;

union FragU { v16h h; unsigned u[8]; };

// ---- WMMA fragment loaders (CDNA5 16-bit layouts, wave32) ----
// A (16x32, MxK): lane l holds row m=l&15; half=l>>4 selects K-subsets.
// packed pair index within row = 8*(v>>2) + 4*half + (v&3)
__device__ __forceinline__ v16h lda_lds(const _Float16* A, int strideH,
                                        int mbase, int kbase, int mmax) {
    int lane = threadIdx.x & 31;
    int m = mbase + (lane & 15);
    int hi = lane >> 4;
    FragU f;
    if (m < mmax) {
        const unsigned* row = (const unsigned*)(A + m * strideH + kbase);
#pragma unroll
        for (int v = 0; v < 8; ++v) {
            int pidx = ((v >> 2) << 3) + (hi << 2) + (v & 3);
            f.u[v] = row[pidx];
        }
    } else {
#pragma unroll
        for (int v = 0; v < 8; ++v) f.u[v] = 0u;
    }
    return f.h;
}

// B (32x16, KxN): lane l holds column n=nbase+(l&15); element s has k = 16*(s>>3) + 8*(l>>4) + (s&7).
__device__ __forceinline__ v16h ldb_glb(const _Float16* BT, int strideN,
                                        int kbase, int nbase) {
    int lane = threadIdx.x & 31;
    int hi = lane >> 4;
    const _Float16* col = BT + (size_t)kbase * strideN + nbase + (lane & 15);
    v16h f;
#pragma unroll
    for (int s = 0; s < 16; ++s) f[s] = col[(size_t)(((s >> 3) << 4) + (hi << 3) + (s & 7)) * strideN];
    return f;
}

__device__ __forceinline__ v8f wmma16(v16h a, v16h b, v8f c) {
    return __builtin_amdgcn_wmma_f32_16x16x32_f16(false, a, false, b, (short)0, c,
                                                  false, false);
}

__device__ __forceinline__ float sigmoidf_(float x) { return 1.f / (1.f + __expf(-x)); }
__device__ __forceinline__ float softplusf_(float x) { return (x > 20.f) ? x : log1pf(__expf(x)); }

// ---- Kernel 0: convert mode-0 weights to f16 transposed (KxN) layouts in ws ----
// ws halves: [0)     w1in_T  64x256
//            [16384) w1xp_T 128x48  (N padded 36->48 with zeros)
//            [22528) w1out_T 128x64
__global__ __launch_bounds__(256) void cvt_weights_kernel(
    const float* __restrict__ in_w,   // (256,64)
    const float* __restrict__ xp_w,   // (36,128)
    const float* __restrict__ out_w,  // (64,128)
    _Float16* __restrict__ ws) {
    int i = blockIdx.x * 256 + threadIdx.x;
    if (i < 64 * 256) {
        int k = i >> 8, n = i & 255;
        ws[i] = (_Float16)in_w[n * 64 + k];
    }
    if (i < 128 * 48) {
        int k = i / 48, n = i % 48;
        ws[16384 + i] = (_Float16)((n < 36) ? xp_w[n * 128 + k] : 0.f);
    }
    if (i < 128 * 64) {
        int k = i >> 6, n = i & 63;
        ws[22528 + i] = (_Float16)out_w[n * 128 + k];
    }
}

// ---- Kernel 1: mode-0 branch. One workgroup (8 waves) per 8x8 block. ----
// L=65 tokens (gt1 + 64 spatial), d_model=64, d_inner=128, d_state=16, dt_rank=4.
__global__ __launch_bounds__(256) void mode0_kernel(
    const float* __restrict__ x, const float* __restrict__ gt1,
    const float* __restrict__ n1w, const float* __restrict__ n1b,
    const float* __restrict__ conv_w, const float* __restrict__ conv_b,
    const float* __restrict__ dt_w, const float* __restrict__ dt_b,
    const float* __restrict__ Alog, const float* __restrict__ Dp,
    const _Float16* __restrict__ w1in_T,   // 64x256
    const _Float16* __restrict__ w1xp_T,   // 128x48
    const _Float16* __restrict__ w1out_T,  // 128x64
    float* __restrict__ out) {
    // LDS arena (phase-overlapped):
    //  [0,16900)      P1: blk f32 65 rows x64, stride 65 (row 64 = gt1)
    //  [0,16384)      P2+: z16 64x128 f16
    //  [16640,33280)  P2: xi16 65x128 f16          / P5+: ygate 64x128 f16
    //  [33280,43520)  P1-P2: aln f16 80x64         / P3+: xc16 65x128 f16
    //  [49920,62400)  P4+: xdbl f32 65x48
    __shared__ __align__(16) char smem[62400];
    float*    blk  = (float*)smem;                   // token-major, stride 65
    _Float16* z16  = (_Float16*)smem;                // 64x128
    _Float16* xi16 = (_Float16*)(smem + 16640);      // 65x128
    _Float16* yg16 = (_Float16*)(smem + 16640);      // 64x128 (after conv)
    _Float16* aln  = (_Float16*)(smem + 33280);      // 80x64
    _Float16* xc16 = (_Float16*)(smem + 33280);      // 65x128 (after GEMM1)
    float*    xdbl = (float*)(smem + 49920);         // 65x48

    int bid = blockIdx.x;
    int bi = bid >> 8;
    int rem = bid & 255;
    int bh = rem >> 4, bw = rem & 15;
    const float* xblk = x + (size_t)bi * 64 * 128 * 128 + (size_t)bh * 8 * 128 + (size_t)bw * 8;
    float* oblk = out + (size_t)bi * 64 * 128 * 128 + (size_t)bh * 8 * 128 + (size_t)bw * 8;

    // ---- P1a: stage block via float4 into token-major LDS (stride 65) ----
    for (int idx = threadIdx.x; idx < 1024; idx += 256) {
        int c = idx >> 4, q = idx & 15;
        int i = q >> 1, jq = (q & 1) << 2;
        const float4 v = *(const float4*)(xblk + c * 16384 + i * 128 + jq);
        int p = i * 8 + jq;
        blk[(p + 0) * 65 + c] = v.x;
        blk[(p + 1) * 65 + c] = v.y;
        blk[(p + 2) * 65 + c] = v.z;
        blk[(p + 3) * 65 + c] = v.w;
    }
    if (threadIdx.x < 64) blk[64 * 65 + threadIdx.x] = gt1[threadIdx.x];  // gt row
    __syncthreads();

    // ---- P1b: LayerNorm per token -> aln f16 (rows 65..79 zeroed) ----
    {
        int t = threadIdx.x;
        if (t < 65) {
            const float* src = &blk[((t == 0) ? 64 : (t - 1)) * 65];
            float s = 0.f, s2 = 0.f;
#pragma unroll 8
            for (int c = 0; c < 64; ++c) { float v = src[c]; s += v; s2 += v * v; }
            float m = s * (1.f / 64.f);
            float var = s2 * (1.f / 64.f) - m * m;
            float inv = rsqrtf(var + 1e-5f);
#pragma unroll 8
            for (int c = 0; c < 64; ++c)
                aln[t * 64 + c] = (_Float16)(((src[c] - m) * inv) * n1w[c] + n1b[c]);
        } else if (t < 80) {
#pragma unroll 8
            for (int c = 0; c < 64; ++c) aln[t * 64 + c] = (_Float16)0.f;
        }
    }
    __syncthreads();

    int wave = threadIdx.x >> 5;
    int lane = threadIdx.x & 31;
    int nl = lane & 15;
    int mh = (lane >> 4) << 3;

    // ---- P2: GEMM1 xz = LN(s0) @ in_w^T : (80x64)x(64x256), WMMA f16 ----
    for (int tt = wave; tt < 80; tt += 8) {
        int mt = tt >> 4, nt = tt & 15;
        int mbase = mt * 16, nbase = nt * 16;
        v8f acc = {};
#pragma unroll
        for (int kk = 0; kk < 64; kk += 32) {
            v16h a = lda_lds(aln, 64, mbase, kk, 80);
            v16h b = ldb_glb(w1in_T, 256, kk, nbase);
            acc = wmma16(a, b, acc);
        }
        // tile-uniform store selection (n<128 decided by nbase; m guards only on edge tiles)
        if (nbase < 128) {
            _Float16* dst = xi16 + nbase + nl;
            if (mbase < 64) {
#pragma unroll
                for (int r = 0; r < 8; ++r)
                    dst[(mbase + mh + r) * 128] = (_Float16)acc[r];
            } else if (mh == 0) {
                dst[64 * 128] = (_Float16)acc[0];     // only m=64 valid
            }
        } else {
            _Float16* dst = z16 + (nbase - 128) + nl;
            if (mbase == 0) {
#pragma unroll
                for (int r = 0; r < 8; ++r) {
                    int m = mh + r;
                    if (m >= 1) dst[(m - 1) * 128] = (_Float16)acc[r];
                }
            } else if (mbase < 64) {
#pragma unroll
                for (int r = 0; r < 8; ++r)
                    dst[(mbase + mh + r - 1) * 128] = (_Float16)acc[r];
            } else if (mh == 0) {
                dst[63 * 128] = (_Float16)acc[0];     // m=64 -> z row 63
            }
        }
    }
    __syncthreads();

    // ---- P3: causal depthwise conv (w=4) + SiLU -> xc16 ----
    for (int idx = threadIdx.x; idx < 65 * 128; idx += 256) {
        int t = idx >> 7, d = idx & 127;
        float acc = conv_b[d];
#pragma unroll
        for (int k = 0; k < 4; ++k) {
            int ts = t - 3 + k;
            if (ts >= 0) acc += conv_w[d * 4 + k] * (float)xi16[ts * 128 + d];
        }
        xc16[t * 128 + d] = (_Float16)(acc * sigmoidf_(acc));
    }
    __syncthreads();

    // ---- P4: GEMM2 x_dbl = xc @ xp_w^T : (80x128)x(128x48), WMMA f16 ----
    for (int tt = wave; tt < 15; tt += 8) {
        int mt = tt / 3, nt = tt % 3;
        int mbase = mt * 16, nbase = nt * 16;
        v8f acc = {};
#pragma unroll
        for (int kk = 0; kk < 128; kk += 32) {
            v16h a = lda_lds(xc16, 128, mbase, kk, 65);  // rows>=65 -> zeros
            v16h b = ldb_glb(w1xp_T, 48, kk, nbase);
            acc = wmma16(a, b, acc);
        }
        float* dst = xdbl + nbase + nl;
        if (mbase < 64) {
#pragma unroll
            for (int r = 0; r < 8; ++r)
                dst[(mbase + mh + r) * 48] = acc[r];
        } else if (mh == 0) {
            dst[64 * 48] = acc[0];
        }
    }
    __syncthreads();

    // ---- P5: selective scan. Thread d (0..127) owns state row h[d][0..15]. ----
    if (threadIdx.x < 128) {
        int d = threadIdx.x;
        float dtw0 = dt_w[d * 4 + 0], dtw1 = dt_w[d * 4 + 1];
        float dtw2 = dt_w[d * 4 + 2], dtw3 = dt_w[d * 4 + 3];
        float dtbd = dt_b[d];
        float Dd = Dp[d];
        float Av[16], h[16];
#pragma unroll
        for (int s = 0; s < 16; ++s) { Av[s] = -__expf(Alog[d * 16 + s]); h[s] = 0.f; }
        for (int t = 0; t < 65; ++t) {
            const float* row = &xdbl[t * 48];
            float4 dl = *(const float4*)row;
            float dt = softplusf_(dl.x * dtw0 + dl.y * dtw1 + dl.z * dtw2 + dl.w * dtw3 + dtbd);
            float xt = (float)xc16[t * 128 + d];
            float dtxt = dt * xt;
            float y = 0.f;
#pragma unroll
            for (int s = 0; s < 16; ++s) {
                float hb = __expf(dt * Av[s]) * h[s] + dtxt * row[4 + s];
                h[s] = hb;
                y += hb * row[20 + s];
            }
            if (t >= 1) {
                float zt = (float)z16[(t - 1) * 128 + d];
                float g = zt * sigmoidf_(zt);
                yg16[(t - 1) * 128 + d] = (_Float16)((y + xt * Dd) * g);
            }
        }
    }
    __syncthreads();

    // ---- P6: GEMM3 o0 = ygate @ out_w^T : (64x128)x(128x64), write o0/3 ----
    // Per lane the 8 accumulators map to 8 consecutive tokens (one image row of
    // the block) -> two contiguous float4 global stores.
    for (int tt = wave; tt < 16; tt += 8) {
        int mt = tt >> 2, nt = tt & 3;
        int mbase = mt * 16, nbase = nt * 16;
        v8f acc = {};
#pragma unroll
        for (int kk = 0; kk < 128; kk += 32) {
            v16h a = lda_lds(yg16, 128, mbase, kk, 64);
            v16h b = ldb_glb(w1out_T, 64, kk, nbase);
            acc = wmma16(a, b, acc);
        }
        int n = nbase + nl;                 // channel c
        int mrow = mbase + mh;              // token base (multiple of 8)
        int i = mrow >> 3;                  // block row; j = r (contiguous)
        float4 lo, hi;
        lo.x = acc[0] * (1.f / 3.f); lo.y = acc[1] * (1.f / 3.f);
        lo.z = acc[2] * (1.f / 3.f); lo.w = acc[3] * (1.f / 3.f);
        hi.x = acc[4] * (1.f / 3.f); hi.y = acc[5] * (1.f / 3.f);
        hi.z = acc[6] * (1.f / 3.f); hi.w = acc[7] * (1.f / 3.f);
        float4* dst = (float4*)(oblk + (size_t)n * 16384 + (size_t)i * 128);
        dst[0] = lo;
        dst[1] = hi;
    }
}

// ---- Kernel 2: modes 1/2 (feature dim 8, d_inner 16, L=513). VALU path. ----
__global__ __launch_bounds__(256) void mode12_kernel(
    const float* __restrict__ x, const float* __restrict__ gt2,
    const float* __restrict__ n2w, const float* __restrict__ n2b,
    const float* __restrict__ in_w,   // (32,8)
    const float* __restrict__ conv_w, const float* __restrict__ conv_b,  // (16,4),(16)
    const float* __restrict__ xp_w,   // (33,16)
    const float* __restrict__ dt_w, const float* __restrict__ dt_b,      // (16,1),(16)
    const float* __restrict__ Alog, const float* __restrict__ Dp,        // (16,16),(16)
    const float* __restrict__ out_w,  // (8,16)
    float* __restrict__ out, int mode) {
    // LDS: [0,16416)  P1: blk2 f32 64x64 (c-major) / P2+: xc16 513x16 f16
    //      [16416,32800) z16 512x16
    //      [32800,49216) xi16 513x16 (reused as ygate 512x16 after conv)
    //      [49216,50240) in_w | [50240,52352) xp_w | [52352,52864) out_w
    __shared__ __align__(16) char smem[52864];
    float*    blk2 = (float*)smem;
    _Float16* xc16 = (_Float16*)smem;
    _Float16* z16  = (_Float16*)(smem + 16416);
    _Float16* xi16 = (_Float16*)(smem + 32800);
    _Float16* yg16 = xi16;
    float* pin  = (float*)(smem + 49216);
    float* pxp  = (float*)(smem + 50240);
    float* pout = (float*)(smem + 52352);

    int bid = blockIdx.x;
    int bi = bid >> 8;
    int rem = bid & 255;
    int bh = rem >> 4, bw = rem & 15;
    const float* xblk = x + (size_t)bi * 64 * 128 * 128 + (size_t)bh * 8 * 128 + (size_t)bw * 8;
    float* oblk = out + (size_t)bi * 64 * 128 * 128 + (size_t)bh * 8 * 128 + (size_t)bw * 8;

    // stage small weights + block (coalesced float4)
    pin[threadIdx.x] = in_w[threadIdx.x];
    for (int i = threadIdx.x; i < 528; i += 256) pxp[i] = xp_w[i];
    if (threadIdx.x < 128) pout[threadIdx.x] = out_w[threadIdx.x];
    for (int idx = threadIdx.x; idx < 1024; idx += 256) {
        int c = idx >> 4, q = idx & 15;
        int i = q >> 1, jq = (q & 1) << 2;
        const float4 v = *(const float4*)(xblk + c * 16384 + i * 128 + jq);
        *(float4*)(blk2 + idx * 4) = v;   // blk2[c*64 + i*8 + j]
    }
    __syncthreads();

    // ---- P1: LN + in_proj per token (reads LDS block) ----
    for (int t = threadIdx.x; t < 513; t += 256) {
        float f[8];
        if (t == 0) {
#pragma unroll
            for (int ff = 0; ff < 8; ++ff) f[ff] = gt2[ff];
        } else {
            int p = t - 1;
            if (mode == 1) {              // token (c,j), feature i
                int c = p >> 3, j = p & 7;
#pragma unroll
                for (int ff = 0; ff < 8; ++ff) f[ff] = blk2[c * 64 + ff * 8 + j];
            } else {                      // token (i,c), feature j
                int i = p >> 6, c = p & 63;
                float4 a = *(const float4*)(blk2 + c * 64 + i * 8);
                float4 b = *(const float4*)(blk2 + c * 64 + i * 8 + 4);
                f[0] = a.x; f[1] = a.y; f[2] = a.z; f[3] = a.w;
                f[4] = b.x; f[5] = b.y; f[6] = b.z; f[7] = b.w;
            }
        }
        float s = 0.f, s2 = 0.f;
#pragma unroll
        for (int ff = 0; ff < 8; ++ff) { s += f[ff]; s2 += f[ff] * f[ff]; }
        float m = s * 0.125f;
        float var = s2 * 0.125f - m * m;
        float inv = rsqrtf(var + 1e-5f);
        float ln[8];
#pragma unroll
        for (int ff = 0; ff < 8; ++ff) ln[ff] = (f[ff] - m) * inv * n2w[ff] + n2b[ff];
#pragma unroll 4
        for (int n = 0; n < 32; ++n) {
            float acc = 0.f;
#pragma unroll
            for (int ff = 0; ff < 8; ++ff) acc += ln[ff] * pin[n * 8 + ff];
            if (n < 16) xi16[t * 16 + n] = (_Float16)acc;
            else if (t >= 1) z16[(t - 1) * 16 + (n - 16)] = (_Float16)acc;
        }
    }
    __syncthreads();

    // ---- P2: causal conv + SiLU -> xc16 (overwrites blk2, now dead) ----
    for (int idx = threadIdx.x; idx < 513 * 16; idx += 256) {
        int t = idx >> 4, d = idx & 15;
        float acc = conv_b[d];
#pragma unroll
        for (int k = 0; k < 4; ++k) {
            int ts = t - 3 + k;
            if (ts >= 0) acc += conv_w[d * 4 + k] * (float)xi16[ts * 16 + d];
        }
        xc16[t * 16 + d] = (_Float16)(acc * sigmoidf_(acc));
    }
    __syncthreads();

    // ---- P3: scan. thread = (d = tid>>4, s = tid&15); shfl-reduce over s. ----
    {
        int d = threadIdx.x >> 4, s = threadIdx.x & 15;
        float Ads = -__expf(Alog[d * 16 + s]);
        float dtwd = dt_w[d];
        float dtbd = dt_b[d];
        float Dd = Dp[d];
        float h = 0.f;
        for (int t = 0; t < 513; ++t) {
            float xr[16];
#pragma unroll
            for (int k = 0; k < 16; ++k) xr[k] = (float)xc16[t * 16 + k];
            float xd0 = 0.f, Bt = 0.f, Ct = 0.f;
#pragma unroll
            for (int k = 0; k < 16; ++k) {
                xd0 += xr[k] * pxp[k];                    // x_dbl col 0 (dt)
                Bt  += xr[k] * pxp[(1 + s) * 16 + k];     // B_t[s]
                Ct  += xr[k] * pxp[(17 + s) * 16 + k];    // C_t[s]
            }
            float dt = softplusf_(xd0 * dtwd + dtbd);
            float xt = xr[d];
            h = __expf(dt * Ads) * h + dt * xt * Bt;
            float yc = h * Ct;
            yc += __shfl_xor(yc, 1, 16);
            yc += __shfl_xor(yc, 2, 16);
            yc += __shfl_xor(yc, 4, 16);
            yc += __shfl_xor(yc, 8, 16);
            if (s == 0 && t >= 1) {
                float zt = (float)z16[(t - 1) * 16 + d];
                float g = zt * sigmoidf_(zt);
                yg16[(t - 1) * 16 + d] = (_Float16)((yc + xt * Dd) * g);
            }
        }
    }
    __syncthreads();

    // ---- P4: out_proj; each thread owns one contiguous 8-float output row ----
    for (int idx = threadIdx.x; idx < 512; idx += 256) {
        int c = idx >> 3, i = idx & 7;
        float o8[8];
        if (mode == 1) {                  // element (c,i,j) <- token (c,j), feat i
#pragma unroll
            for (int j = 0; j < 8; ++j) {
                int p = c * 8 + j;
                float a = 0.f;
#pragma unroll
                for (int d = 0; d < 16; ++d) a += (float)yg16[p * 16 + d] * pout[i * 16 + d];
                o8[j] = a;
            }
        } else {                          // element (c,i,j) <- token (i,c), feat j
            int p = i * 64 + c;
            float yv[16];
#pragma unroll
            for (int d = 0; d < 16; ++d) yv[d] = (float)yg16[p * 16 + d];
#pragma unroll
            for (int j = 0; j < 8; ++j) {
                float a = 0.f;
#pragma unroll
                for (int d = 0; d < 16; ++d) a += yv[d] * pout[j * 16 + d];
                o8[j] = a;
            }
        }
        float4* dst = (float4*)(oblk + (size_t)c * 16384 + (size_t)i * 128);
        float4 d0 = dst[0], d1 = dst[1];
        d0.x += o8[0] * (1.f / 3.f); d0.y += o8[1] * (1.f / 3.f);
        d0.z += o8[2] * (1.f / 3.f); d0.w += o8[3] * (1.f / 3.f);
        d1.x += o8[4] * (1.f / 3.f); d1.y += o8[5] * (1.f / 3.f);
        d1.z += o8[6] * (1.f / 3.f); d1.w += o8[7] * (1.f / 3.f);
        dst[0] = d0;
        dst[1] = d1;
    }
}

extern "C" void kernel_launch(void* const* d_in, const int* in_sizes, int n_in,
                              void* d_out, int out_size, void* d_ws, size_t ws_size,
                              hipStream_t stream) {
    const float* x        = (const float*)d_in[0];
    const float* gt1      = (const float*)d_in[1];
    const float* gt2      = (const float*)d_in[2];
    const float* n1_w     = (const float*)d_in[3];
    const float* n1_b     = (const float*)d_in[4];
    const float* n2_w     = (const float*)d_in[5];
    const float* n2_b     = (const float*)d_in[6];
    const float* m1_in_w  = (const float*)d_in[7];
    const float* m1_conv_w= (const float*)d_in[8];
    const float* m1_conv_b= (const float*)d_in[9];
    const float* m1_xp_w  = (const float*)d_in[10];
    const float* m1_dt_w  = (const float*)d_in[11];
    const float* m1_dt_b  = (const float*)d_in[12];
    const float* m1_Alog  = (const float*)d_in[13];
    const float* m1_D     = (const float*)d_in[14];
    const float* m1_out_w = (const float*)d_in[15];
    const float* m2_in_w  = (const float*)d_in[16];
    const float* m2_conv_w= (const float*)d_in[17];
    const float* m2_conv_b= (const float*)d_in[18];
    const float* m2_xp_w  = (const float*)d_in[19];
    const float* m2_dt_w  = (const float*)d_in[20];
    const float* m2_dt_b  = (const float*)d_in[21];
    const float* m2_Alog  = (const float*)d_in[22];
    const float* m2_D     = (const float*)d_in[23];
    const float* m2_out_w = (const float*)d_in[24];
    float* out = (float*)d_out;
    _Float16* wsH = (_Float16*)d_ws;

    // 1) weight conversion (f16 transposed layouts in ws)
    cvt_weights_kernel<<<64, 256, 0, stream>>>(m1_in_w, m1_xp_w, m1_out_w, wsH);

    // 2) mode-0 branch: writes o0/3 (full overwrite of d_out)
    mode0_kernel<<<512, 256, 0, stream>>>(
        x, gt1, n1_w, n1_b, m1_conv_w, m1_conv_b, m1_dt_w, m1_dt_b,
        m1_Alog, m1_D, wsH, wsH + 16384, wsH + 22528, out);

    // 3) mode-1 then mode-2 branches: sequential stream order -> deterministic +=
    mode12_kernel<<<512, 256, 0, stream>>>(
        x, gt2, n2_w, n2_b, m2_in_w, m2_conv_w, m2_conv_b, m2_xp_w,
        m2_dt_w, m2_dt_b, m2_Alog, m2_D, m2_out_w, out, 1);
    mode12_kernel<<<512, 256, 0, stream>>>(
        x, gt2, n2_w, n2_b, m2_in_w, m2_conv_w, m2_conv_b, m2_xp_w,
        m2_dt_w, m2_dt_b, m2_Alog, m2_D, m2_out_w, out, 2);
}
